// ContactPositionRegressor_34102040330514
// MI455X (gfx1250) — compile-verified, matrix operand order checked by reading, not run on hardware
//
#include <hip/hip_runtime.h>
#include <hip/hip_bf16.h>

// ---------------------------------------------------------------------------
// Problem constants (match reference)
// ---------------------------------------------------------------------------
#define B_  8
#define N_  512
#define M_  1024
#define C_  512
#define H_  8
#define DH_ 64
#define EMB_ 32

typedef __attribute__((ext_vector_type(16))) __bf16 v16bf;
typedef __attribute__((ext_vector_type(8)))  __bf16 v8bf;
typedef __attribute__((ext_vector_type(4)))  __bf16 v4bf;
typedef __attribute__((ext_vector_type(8)))  float  v8f;

// ---------------------------------------------------------------------------
// WMMA helpers (CDNA5: v_wmma_f32_16x16x32_bf16, wave32)
// A-fragment (16-bit, 16x32): lane%16 = row, lane/16 = K-half; per lane two
// contiguous 8-element runs at k = kh*8 and k = 16 + kh*8.  B fragments use
// the same pattern on B^T (col-major staged).  C/D: VGPR r -> row
// (lane/16)*8 + r, col = lane%16.
// ---------------------------------------------------------------------------
__device__ __forceinline__ v8f wmma_bf16(v16bf a, v16bf b, v8f c) {
  return __builtin_amdgcn_wmma_f32_16x16x32_bf16(
      /*neg_a=*/false, a, /*neg_b=*/false, b,
      /*c_mod=*/(short)0, c, /*reuse_a=*/false, /*reuse_b=*/false);
}

__device__ __forceinline__ v16bf ld_frag(const __bf16* base, int stride, int lane) {
  const int m  = lane & 15;
  const int kh = lane >> 4;
  const __bf16* p = base + m * stride + kh * 8;
  union { v16bf v; v8bf h[2]; } u;
  u.h[0] = *reinterpret_cast<const v8bf*>(p);
  u.h[1] = *reinterpret_cast<const v8bf*>(p + 16);
  return u.v;
}

// Pack 4 fp32 -> 4 bf16, single 8-byte LDS store (ds_store_b64).
__device__ __forceinline__ void st4_bf16(__bf16* dst, float4 f) {
  v4bf x;
  x[0] = (__bf16)f.x; x[1] = (__bf16)f.y; x[2] = (__bf16)f.z; x[3] = (__bf16)f.w;
  *reinterpret_cast<v4bf*>(dst) = x;
}

__device__ __forceinline__ v4bf pack4(float a, float b, float c, float d) {
  v4bf x;
  x[0] = (__bf16)a; x[1] = (__bf16)b; x[2] = (__bf16)c; x[3] = (__bf16)d;
  return x;
}

// ---------------------------------------------------------------------------
// Generic GEMM:  Out[R,N] = act(A[R,K] @ W[K,N] + bias [+ Res])
// fp32 in/out, bf16 WMMA fp32-accumulate.  64x64 tile, 4 waves, K-step 64.
// K, N, RES, RELU are compile-time: unrolled epilogue with immediate offsets.
// ---------------------------------------------------------------------------
template <int K, int N, bool RES, bool RELU>
__global__ __launch_bounds__(128)
void gemm_bf16_wmma(const float* __restrict__ A, const float* __restrict__ W,
                    const float* __restrict__ bias, const float* __restrict__ Res,
                    float* __restrict__ Out) {
  __shared__ __bf16 As[64][64];   // A tile, row-major          (8 KB)
  __shared__ __bf16 Wt[64][64];   // W tile transposed [n][k]   (8 KB)

  const int t    = threadIdx.x;
  const int lane = t & 31;
  const int w    = t >> 5;
  const int row0 = blockIdx.x * 64;
  const int n0   = blockIdx.y * 64;
  const int wr   = (w >> 1) * 32;
  const int wc   = (w & 1)  * 32;

  v8f acc[2][2] = {};

  for (int k0 = 0; k0 < K; k0 += 64) {
    // Stage A: 64x64 fp32 -> bf16, packed b64 stores.
#pragma unroll
    for (int i = 0; i < 8; ++i) {
      const int idx = t + i * 128;
      const int r   = idx >> 4;
      const int c4  = (idx & 15) << 2;
      const float4 f = *reinterpret_cast<const float4*>(
          A + (size_t)(row0 + r) * K + k0 + c4);
      st4_bf16(&As[r][c4], f);
    }
    // Stage W transposed via 4x4 register micro-transpose blocks.
#pragma unroll
    for (int i = 0; i < 2; ++i) {
      const int idx = t + i * 128;        // 256 blocks of 4k x 4n
      const int kb  = (idx >> 4) << 2;
      const int cb  = (idx & 15) << 2;
      const float4 f0 = *reinterpret_cast<const float4*>(W + (size_t)(k0 + kb + 0) * N + n0 + cb);
      const float4 f1 = *reinterpret_cast<const float4*>(W + (size_t)(k0 + kb + 1) * N + n0 + cb);
      const float4 f2 = *reinterpret_cast<const float4*>(W + (size_t)(k0 + kb + 2) * N + n0 + cb);
      const float4 f3 = *reinterpret_cast<const float4*>(W + (size_t)(k0 + kb + 3) * N + n0 + cb);
      *reinterpret_cast<v4bf*>(&Wt[cb + 0][kb]) = pack4(f0.x, f1.x, f2.x, f3.x);
      *reinterpret_cast<v4bf*>(&Wt[cb + 1][kb]) = pack4(f0.y, f1.y, f2.y, f3.y);
      *reinterpret_cast<v4bf*>(&Wt[cb + 2][kb]) = pack4(f0.z, f1.z, f2.z, f3.z);
      *reinterpret_cast<v4bf*>(&Wt[cb + 3][kb]) = pack4(f0.w, f1.w, f2.w, f3.w);
    }
    // Prefetch next K-tile (global_prefetch_b8 on gfx1250).
    if (k0 + 64 < K) {
      __builtin_prefetch(A + (size_t)(row0 + (t >> 1)) * K + k0 + 64, 0, 0);
      __builtin_prefetch(W + (size_t)(k0 + 64 + (t >> 1)) * N + n0, 0, 0);
    }
    __syncthreads();

#pragma unroll
    for (int ks = 0; ks < 2; ++ks) {
      const v16bf a0 = ld_frag(&As[wr     ][ks * 32], 64, lane);
      const v16bf a1 = ld_frag(&As[wr + 16][ks * 32], 64, lane);
      const v16bf b0 = ld_frag(&Wt[wc     ][ks * 32], 64, lane);
      const v16bf b1 = ld_frag(&Wt[wc + 16][ks * 32], 64, lane);
      acc[0][0] = wmma_bf16(a0, b0, acc[0][0]);
      acc[0][1] = wmma_bf16(a0, b1, acc[0][1]);
      acc[1][0] = wmma_bf16(a1, b0, acc[1][0]);
      acc[1][1] = wmma_bf16(a1, b1, acc[1][1]);
    }
    __syncthreads();
  }

  // Epilogue: fully unrolled, constant strides.
  const int rbase = row0 + wr + ((lane >> 4) << 3);
  const int cbase = n0 + wc + (lane & 15);
#pragma unroll
  for (int ni = 0; ni < 2; ++ni) {
    const int col = cbase + ni * 16;
    const float bv = bias[col];
#pragma unroll
    for (int mi = 0; mi < 2; ++mi) {
      const int row = rbase + mi * 16;
      float* __restrict__ op = Out + (size_t)row * N + col;
      const float* __restrict__ rp = RES ? (Res + (size_t)row * N + col) : nullptr;
#pragma unroll
      for (int r8 = 0; r8 < 8; ++r8) {
        float v = acc[mi][ni][r8] + bv;
        if (RES)  v += rp[r8 * N];
        if (RELU) v = fmaxf(v, 0.0f);
        op[r8 * N] = v;
      }
    }
  }
}

// ---------------------------------------------------------------------------
// Flash cross-attention: one workgroup per (b, h, 64-query tile).
// S = (Q*0.125) @ K^T via WMMA, online softmax in LDS, O += P @ V via WMMA.
// ---------------------------------------------------------------------------
__global__ __launch_bounds__(128)
void flash_attn_wmma(const float* __restrict__ Q, const float* __restrict__ Kb,
                     const float* __restrict__ Vb, float* __restrict__ O) {
  __shared__ __bf16 Qs [64][64];   // scaled queries                  (8 KB)
  __shared__ __bf16 KVt[64][64];   // K tile [key][dh] / V^T [d][key] (8 KB)
  __shared__ float  Sst[64][64];   // scores, TRANSPOSED: [key][q]    (16 KB)
  __shared__ __bf16 Ps [64][64];   // softmax numerators [q][key]     (8 KB)
  __shared__ float  row_m[64], row_l[64], row_c[64];

  const int t    = threadIdx.x;
  const int lane = t & 31;
  const int w    = t >> 5;
  const int wr   = (w >> 1) * 32;
  const int wc   = (w & 1)  * 32;
  const int bb   = blockIdx.z;
  const int h    = blockIdx.y;
  const int n0   = blockIdx.x * 64;

  const size_t qbase  = ((size_t)bb * N_ + n0) * C_ + (size_t)h * DH_;
  const size_t kvbase = ((size_t)bb * M_) * C_ + (size_t)h * DH_;

  // Stage Q tile (scale 1/sqrt(DH)=0.125 folded into bf16).
#pragma unroll
  for (int i = 0; i < 8; ++i) {
    const int idx = t + i * 128;
    const int r   = idx >> 4;
    const int c4  = (idx & 15) << 2;
    const float4 f = *reinterpret_cast<const float4*>(Q + qbase + (size_t)r * C_ + c4);
    *reinterpret_cast<v4bf*>(&Qs[r][c4]) =
        pack4(f.x * 0.125f, f.y * 0.125f, f.z * 0.125f, f.w * 0.125f);
  }
  if (t < 64) { row_m[t] = -1e30f; row_l[t] = 0.0f; }
  __syncthreads();

  // Hoist Q fragments: [mi][kstep], kstep over dh=64.
  v16bf qa[2][2];
#pragma unroll
  for (int mi = 0; mi < 2; ++mi)
#pragma unroll
    for (int ks = 0; ks < 2; ++ks)
      qa[mi][ks] = ld_frag(&Qs[wr + mi * 16][ks * 32], 64, lane);

  v8f o[2][2] = {};

  for (int m0 = 0; m0 < M_; m0 += 64) {
    // Stage K tile [key][dh], packed stores.
#pragma unroll
    for (int i = 0; i < 8; ++i) {
      const int idx = t + i * 128;
      const int r   = idx >> 4;
      const int c4  = (idx & 15) << 2;
      const float4 f = *reinterpret_cast<const float4*>(
          Kb + kvbase + (size_t)(m0 + r) * C_ + c4);
      st4_bf16(&KVt[r][c4], f);
    }
    __syncthreads();

    // S tile: 64q x 64keys; each wave 32x32.
    v8f s[2][2] = {};
#pragma unroll
    for (int ks = 0; ks < 2; ++ks) {
      const v16bf kb0 = ld_frag(&KVt[wc     ][ks * 32], 64, lane);
      const v16bf kb1 = ld_frag(&KVt[wc + 16][ks * 32], 64, lane);
      s[0][0] = wmma_bf16(qa[0][ks], kb0, s[0][0]);
      s[0][1] = wmma_bf16(qa[0][ks], kb1, s[0][1]);
      s[1][0] = wmma_bf16(qa[1][ks], kb0, s[1][0]);
      s[1][1] = wmma_bf16(qa[1][ks], kb1, s[1][1]);
    }
    // Store S transposed: accumulator rows are contiguous -> one v8f store.
#pragma unroll
    for (int mi = 0; mi < 2; ++mi)
#pragma unroll
      for (int ni = 0; ni < 2; ++ni)
        *reinterpret_cast<v8f*>(
            &Sst[wc + ni * 16 + (lane & 15)][wr + mi * 16 + ((lane >> 4) << 3)]) =
            s[mi][ni];
    __syncthreads();

    // Online softmax update (row q = t, threads 0..63); reads Sst[key][q]
    // (conflict-free: consecutive q across threads).
    if (t < 64) {
      const float mprev = row_m[t];
      float mx = mprev;
      for (int j = 0; j < 64; ++j) mx = fmaxf(mx, Sst[j][t]);
      const float corr = __expf(mprev - mx);
      float sum = 0.0f;
      for (int j4 = 0; j4 < 16; ++j4) {
        float p0 = __expf(Sst[j4 * 4 + 0][t] - mx);
        float p1 = __expf(Sst[j4 * 4 + 1][t] - mx);
        float p2 = __expf(Sst[j4 * 4 + 2][t] - mx);
        float p3 = __expf(Sst[j4 * 4 + 3][t] - mx);
        sum += (p0 + p1) + (p2 + p3);
        *reinterpret_cast<v4bf*>(&Ps[t][j4 * 4]) = pack4(p0, p1, p2, p3);
      }
      row_l[t] = row_l[t] * corr + sum;
      row_m[t] = mx;
      row_c[t] = corr;
    }
    __syncthreads();

    // Rescale running O by per-row correction.
#pragma unroll
    for (int mi = 0; mi < 2; ++mi)
#pragma unroll
      for (int r8 = 0; r8 < 8; ++r8) {
        const float c = row_c[wr + mi * 16 + ((lane >> 4) << 3) + r8];
        o[mi][0][r8] *= c;
        o[mi][1][r8] *= c;
      }

    // Stage V^T [d][key] via 4x4 register micro-transpose (K tile consumed).
#pragma unroll
    for (int i = 0; i < 2; ++i) {
      const int idx = t + i * 128;        // 256 blocks of 4key x 4d
      const int kg  = (idx >> 4) << 2;    // key group
      const int cg  = (idx & 15) << 2;    // dh group
      const float4 f0 = *reinterpret_cast<const float4*>(Vb + kvbase + (size_t)(m0 + kg + 0) * C_ + cg);
      const float4 f1 = *reinterpret_cast<const float4*>(Vb + kvbase + (size_t)(m0 + kg + 1) * C_ + cg);
      const float4 f2 = *reinterpret_cast<const float4*>(Vb + kvbase + (size_t)(m0 + kg + 2) * C_ + cg);
      const float4 f3 = *reinterpret_cast<const float4*>(Vb + kvbase + (size_t)(m0 + kg + 3) * C_ + cg);
      *reinterpret_cast<v4bf*>(&KVt[cg + 0][kg]) = pack4(f0.x, f1.x, f2.x, f3.x);
      *reinterpret_cast<v4bf*>(&KVt[cg + 1][kg]) = pack4(f0.y, f1.y, f2.y, f3.y);
      *reinterpret_cast<v4bf*>(&KVt[cg + 2][kg]) = pack4(f0.z, f1.z, f2.z, f3.z);
      *reinterpret_cast<v4bf*>(&KVt[cg + 3][kg]) = pack4(f0.w, f1.w, f2.w, f3.w);
    }
    __syncthreads();

    // O += P @ V  (k = 64 keys, two 32-k steps)
#pragma unroll
    for (int ks = 0; ks < 2; ++ks) {
      const v16bf pa0 = ld_frag(&Ps [wr     ][ks * 32], 64, lane);
      const v16bf pa1 = ld_frag(&Ps [wr + 16][ks * 32], 64, lane);
      const v16bf vb0 = ld_frag(&KVt[wc     ][ks * 32], 64, lane);
      const v16bf vb1 = ld_frag(&KVt[wc + 16][ks * 32], 64, lane);
      o[0][0] = wmma_bf16(pa0, vb0, o[0][0]);
      o[0][1] = wmma_bf16(pa0, vb1, o[0][1]);
      o[1][0] = wmma_bf16(pa1, vb0, o[1][0]);
      o[1][1] = wmma_bf16(pa1, vb1, o[1][1]);
    }
    __syncthreads();
  }

  // Invert l once, then unrolled constant-stride stores.
  if (t < 64) row_l[t] = 1.0f / row_l[t];
  __syncthreads();

  const int rbase = wr + ((lane >> 4) << 3);
#pragma unroll
  for (int mi = 0; mi < 2; ++mi) {
    const int rr0 = rbase + mi * 16;
#pragma unroll
    for (int ni = 0; ni < 2; ++ni) {
      const int cc = wc + ni * 16 + (lane & 15);
      float* __restrict__ op =
          O + ((size_t)bb * N_ + n0 + rr0) * C_ + (size_t)h * DH_ + cc;
#pragma unroll
      for (int r8 = 0; r8 < 8; ++r8)
        op[r8 * C_] = o[mi][ni][r8] * row_l[rr0 + r8];
    }
  }
}

// ---------------------------------------------------------------------------
// Small VALU kernels
// ---------------------------------------------------------------------------

// hidden = relu(coord[R,3] @ w1[3,128] + b1); one block per row, 128 threads.
__global__ __launch_bounds__(128)
void posenc_hidden(const float* __restrict__ coord, const float* __restrict__ w1,
                   const float* __restrict__ b1, float* __restrict__ hid) {
  const int row = blockIdx.x;
  const int t   = threadIdx.x;
  const float c0 = coord[row * 3 + 0];
  const float c1 = coord[row * 3 + 1];
  const float c2 = coord[row * 3 + 2];
  const float v  = b1[t] + c0 * w1[t] + c1 * w1[128 + t] + c2 * w1[256 + t];
  hid[(size_t)row * 128 + t] = fmaxf(v, 0.0f);
}

__device__ __forceinline__ float block_sum_256(float v, float* red, int t) {
  red[t] = v; __syncthreads();
  for (int off = 128; off > 0; off >>= 1) {
    if (t < off) red[t] += red[t + off];
    __syncthreads();
  }
  const float r = red[0];
  __syncthreads();
  return r;
}

__device__ __forceinline__ float block_max_256(float v, float* red, int t) {
  red[t] = v; __syncthreads();
  for (int off = 128; off > 0; off >>= 1) {
    if (t < off) red[t] = fmaxf(red[t], red[t + off]);
    __syncthreads();
  }
  const float r = red[0];
  __syncthreads();
  return r;
}

// fused = lf + relu(LN(T) * gamma + beta); one block per row (C=512), 256 thr.
__global__ __launch_bounds__(256)
void fuse_ln_kernel(const float* __restrict__ T, const float* __restrict__ gamma,
                    const float* __restrict__ beta, const float* __restrict__ lf,
                    float* __restrict__ out) {
  __shared__ float red[256];
  const int row = blockIdx.x;
  const int t   = threadIdx.x;
  const float a0 = T[(size_t)row * C_ + t];
  const float a1 = T[(size_t)row * C_ + t + 256];
  const float mean = block_sum_256(a0 + a1, red, t) * (1.0f / (float)C_);
  const float d0 = a0 - mean, d1 = a1 - mean;
  const float var = block_sum_256(d0 * d0 + d1 * d1, red, t) * (1.0f / (float)C_);
  const float rstd = rsqrtf(var + 1e-5f);
  out[(size_t)row * C_ + t] =
      lf[(size_t)row * C_ + t] + fmaxf(d0 * rstd * gamma[t] + beta[t], 0.0f);
  out[(size_t)row * C_ + t + 256] =
      lf[(size_t)row * C_ + t + 256] +
      fmaxf(d1 * rstd * gamma[t + 256] + beta[t + 256], 0.0f);
}

// Attention pooling: per batch (block), logits -> softmax over N -> weighted sum.
__global__ __launch_bounds__(256)
void pool_kernel(const float* __restrict__ hid /*[B*N,128]*/,
                 const float* __restrict__ w2 /*[128]*/, const float* __restrict__ b2,
                 const float* __restrict__ fused /*[B*N,512]*/,
                 float* __restrict__ g /*[B,512]*/) {
  __shared__ float p[N_];
  __shared__ float red[256];
  const int b = blockIdx.x;
  const int t = threadIdx.x;
  for (int n = t; n < N_; n += 256) {
    const float* hr = hid + ((size_t)b * N_ + n) * 128;
    float acc = b2[0];
    for (int j = 0; j < 128; ++j) acc += hr[j] * w2[j];
    p[n] = acc;
  }
  __syncthreads();
  const float mx  = block_max_256(fmaxf(p[t], p[t + 256]), red, t);
  const float e0  = __expf(p[t] - mx);
  const float e1  = __expf(p[t + 256] - mx);
  const float inv = 1.0f / block_sum_256(e0 + e1, red, t);
  __syncthreads();
  p[t] = e0 * inv;
  p[t + 256] = e1 * inv;
  __syncthreads();
  float acc0 = 0.0f, acc1 = 0.0f;
  for (int n = 0; n < N_; ++n) {
    const float* fr = fused + ((size_t)b * N_ + n) * C_;
    const float pn = p[n];
    acc0 += fr[t] * pn;
    acc1 += fr[t + 256] * pn;
  }
  g[(size_t)b * C_ + t] = acc0;
  g[(size_t)b * C_ + t + 256] = acc1;
}

// Regression head: concat cat emb, two LN+ReLU layers, final linear -> [B,3].
__global__ __launch_bounds__(256)
void head_kernel(const float* __restrict__ g, const float* __restrict__ cat_emb,
                 const int* __restrict__ cat,
                 const float* __restrict__ w1, const float* __restrict__ b1,
                 const float* __restrict__ g1, const float* __restrict__ be1,
                 const float* __restrict__ w2, const float* __restrict__ b2,
                 const float* __restrict__ g2, const float* __restrict__ be2,
                 const float* __restrict__ w3, const float* __restrict__ b3,
                 float* __restrict__ out) {
  __shared__ float gx[C_ + EMB_];
  __shared__ float h1[256];
  __shared__ float h2[128];
  __shared__ float red[256];
  const int b = blockIdx.x;
  const int t = threadIdx.x;
  for (int i = t; i < C_; i += 256) gx[i] = g[(size_t)b * C_ + i];
  if (t < EMB_) gx[C_ + t] = cat_emb[cat[b] * EMB_ + t];
  __syncthreads();

  // layer 1: 544 -> 256, LN, ReLU
  float acc = b1[t];
  for (int j = 0; j < C_ + EMB_; ++j) acc += gx[j] * w1[j * 256 + t];
  const float m1 = block_sum_256(acc, red, t) * (1.0f / 256.0f);
  const float d1 = acc - m1;
  const float v1 = block_sum_256(d1 * d1, red, t) * (1.0f / 256.0f);
  h1[t] = fmaxf(d1 * rsqrtf(v1 + 1e-5f) * g1[t] + be1[t], 0.0f);
  __syncthreads();

  // layer 2: 256 -> 128, LN, ReLU  (threads >=128 contribute zeros)
  float acc2 = 0.0f;
  if (t < 128) {
    acc2 = b2[t];
    for (int j = 0; j < 256; ++j) acc2 += h1[j] * w2[j * 128 + t];
  }
  const float m2 = block_sum_256((t < 128) ? acc2 : 0.0f, red, t) * (1.0f / 128.0f);
  const float d2 = acc2 - m2;
  const float v2 = block_sum_256((t < 128) ? d2 * d2 : 0.0f, red, t) * (1.0f / 128.0f);
  if (t < 128) h2[t] = fmaxf(d2 * rsqrtf(v2 + 1e-5f) * g2[t] + be2[t], 0.0f);
  __syncthreads();

  if (t < 3) {
    float o = b3[t];
    for (int j = 0; j < 128; ++j) o += h2[j] * w3[j * 3 + t];
    out[b * 3 + t] = o;
  }
}

// ---------------------------------------------------------------------------
// Launcher
// ---------------------------------------------------------------------------
extern "C" void kernel_launch(void* const* d_in, const int* in_sizes, int n_in,
                              void* d_out, int out_size, void* d_ws, size_t ws_size,
                              hipStream_t stream) {
  (void)in_sizes; (void)n_in; (void)out_size; (void)ws_size;

  const float* local_feat   = (const float*)d_in[0];
  const float* parent_feat  = (const float*)d_in[1];
  const float* local_coord  = (const float*)d_in[2];
  const float* parent_coord = (const float*)d_in[3];
  const int*   category_id  = (const int*)d_in[4];
  const float* lpe_w1 = (const float*)d_in[5];  const float* lpe_b1 = (const float*)d_in[6];
  const float* lpe_w2 = (const float*)d_in[7];  const float* lpe_b2 = (const float*)d_in[8];
  const float* ppe_w1 = (const float*)d_in[9];  const float* ppe_b1 = (const float*)d_in[10];
  const float* ppe_w2 = (const float*)d_in[11]; const float* ppe_b2 = (const float*)d_in[12];
  const float* wq = (const float*)d_in[13]; const float* bq = (const float*)d_in[14];
  const float* wk = (const float*)d_in[15]; const float* bk = (const float*)d_in[16];
  const float* wv = (const float*)d_in[17]; const float* bv = (const float*)d_in[18];
  const float* wo = (const float*)d_in[19]; const float* bo = (const float*)d_in[20];
  const float* fus_w = (const float*)d_in[21]; const float* fus_b    = (const float*)d_in[22];
  const float* fus_g = (const float*)d_in[23]; const float* fus_beta = (const float*)d_in[24];
  const float* ap_w1 = (const float*)d_in[25]; const float* ap_b1 = (const float*)d_in[26];
  const float* ap_w2 = (const float*)d_in[27]; const float* ap_b2 = (const float*)d_in[28];
  const float* cat_emb = (const float*)d_in[29];
  const float* r_w1 = (const float*)d_in[30]; const float* r_b1  = (const float*)d_in[31];
  const float* r_g1 = (const float*)d_in[32]; const float* r_be1 = (const float*)d_in[33];
  const float* r_w2 = (const float*)d_in[34]; const float* r_b2  = (const float*)d_in[35];
  const float* r_g2 = (const float*)d_in[36]; const float* r_be2 = (const float*)d_in[37];
  const float* r_w3 = (const float*)d_in[38]; const float* r_b3  = (const float*)d_in[39];

  // Workspace layout (floats)
  float* ws = (float*)d_ws;
  const size_t SZ_LOC = (size_t)B_ * N_ * C_;   // 2,097,152
  const size_t SZ_PAR = (size_t)B_ * M_ * C_;   // 4,194,304
  float* lf = ws;                    // local feats + PE
  float* pf = lf + SZ_LOC;           // parent feats + PE
  float* qb = pf + SZ_PAR;           // Q  (also PE-hidden / fusion pre-LN scratch)
  float* kb = qb + SZ_LOC;           // K
  float* vb = kb + SZ_PAR;           // V
  float* s1 = vb + SZ_PAR;           // attention out -> fused
  float* s2 = s1 + SZ_LOC;           // out-proj -> pooling hidden
  float* gb = s2 + SZ_LOC;           // pooled global feature [B, C]

  const int RL = B_ * N_;   // 4096
  const int RP = B_ * M_;   // 8192

  // 1) Positional encodings (hidden in qb, then WMMA GEMM with residual add)
  posenc_hidden<<<RL, 128, 0, stream>>>(local_coord, lpe_w1, lpe_b1, qb);
  gemm_bf16_wmma<128, C_, true, false><<<dim3(RL / 64, C_ / 64), 128, 0, stream>>>(
      qb, lpe_w2, lpe_b2, local_feat, lf);
  posenc_hidden<<<RP, 128, 0, stream>>>(parent_coord, ppe_w1, ppe_b1, qb);
  gemm_bf16_wmma<128, C_, true, false><<<dim3(RP / 64, C_ / 64), 128, 0, stream>>>(
      qb, ppe_w2, ppe_b2, parent_feat, pf);

  // 2) Q/K/V projections
  gemm_bf16_wmma<C_, C_, false, false><<<dim3(RL / 64, C_ / 64), 128, 0, stream>>>(
      lf, wq, bq, nullptr, qb);
  gemm_bf16_wmma<C_, C_, false, false><<<dim3(RP / 64, C_ / 64), 128, 0, stream>>>(
      pf, wk, bk, nullptr, kb);
  gemm_bf16_wmma<C_, C_, false, false><<<dim3(RP / 64, C_ / 64), 128, 0, stream>>>(
      pf, wv, bv, nullptr, vb);

  // 3) Flash cross-attention (B*H*8 = 512 workgroups) -> s1
  flash_attn_wmma<<<dim3(N_ / 64, H_, B_), 128, 0, stream>>>(qb, kb, vb, s1);

  // 4) Output projection, fusion projection, LN+ReLU+residual
  gemm_bf16_wmma<C_, C_, false, false><<<dim3(RL / 64, C_ / 64), 128, 0, stream>>>(
      s1, wo, bo, nullptr, s2);
  gemm_bf16_wmma<C_, C_, false, false><<<dim3(RL / 64, C_ / 64), 128, 0, stream>>>(
      s2, fus_w, fus_b, nullptr, qb);
  fuse_ln_kernel<<<RL, 256, 0, stream>>>(qb, fus_g, fus_beta, lf, s1);  // fused in s1

  // 5) Attention pooling: hidden (WMMA GEMM with ReLU), then softmax + pool
  gemm_bf16_wmma<C_, 128, false, true><<<dim3(RL / 64, 128 / 64), 128, 0, stream>>>(
      s1, ap_w1, ap_b1, nullptr, s2);
  pool_kernel<<<B_, 256, 0, stream>>>(s2, ap_w2, ap_b2, s1, gb);

  // 6) Regression head -> d_out [B,3]
  head_kernel<<<B_, 256, 0, stream>>>(gb, cat_emb, category_id,
                                      r_w1, r_b1, r_g1, r_be1,
                                      r_w2, r_b2, r_g2, r_be2,
                                      r_w3, r_b3, (float*)d_out);
}
